// BiRWKVBlock_40157944217756
// MI455X (gfx1250) — matrix operand order repaired and checked by reading, not compiled, not run on hardware
//
#include <hip/hip_runtime.h>
#include <cmath>

// ---------------------------------------------------------------------------
// BiRWKV block for MI455X (gfx1250, wave32, WMMA).
// B=8, T=1024, C=1024.  All GEMMs run through v_wmma_f32_16x16x32_bf16.
// ---------------------------------------------------------------------------

typedef __bf16 bf16;
typedef __attribute__((ext_vector_type(8)))  __bf16 v8bf;
typedef __attribute__((ext_vector_type(16))) __bf16 v16bf;
typedef __attribute__((ext_vector_type(8)))  float  v8f;

static constexpr int Bc = 8;
static constexpr int Tc = 1024;
static constexpr int Cc = 1024;
static constexpr int MT = Bc * Tc;          // 8192 rows
static constexpr int LDT = 40;              // LDS row stride (bf16) for 32-wide tiles, 80B rows

// Epilogue selector
enum { EPI_STORE = 0, EPI_SIGMOID = 1, EPI_RESADD = 2, EPI_RELUSQ = 3, EPI_FINAL = 4 };

__device__ __forceinline__ float sigmoidf_fast(float x) {
    return 1.0f / (1.0f + __expf(-x));
}

// ---------------------------------------------------------------------------
// fp32 -> bf16 conversion (weights & activations)
// ---------------------------------------------------------------------------
__global__ __launch_bounds__(256) void f2bf_kernel(const float* __restrict__ in,
                                                   bf16* __restrict__ out, size_t n) {
    size_t i = (size_t)blockIdx.x * 256u + threadIdx.x;
    if (i < n) out[i] = (bf16)in[i];
}

// ---------------------------------------------------------------------------
// LayerNorm over C=1024, one row per block (256 threads = 8 waves), bf16 out.
// ---------------------------------------------------------------------------
__global__ __launch_bounds__(256) void ln_kernel(const float* __restrict__ x,
                                                 const float* __restrict__ w,
                                                 const float* __restrict__ b,
                                                 bf16* __restrict__ y) {
    __shared__ float red[16];
    const size_t row = blockIdx.x;
    const float* xr = x + row * Cc;
    float s = 0.f, ss = 0.f;
    #pragma unroll
    for (int c = threadIdx.x; c < Cc; c += 256) {
        float v = xr[c];
        s += v;
        ss += v * v;
    }
    #pragma unroll
    for (int o = 16; o > 0; o >>= 1) {          // wave32 butterfly reduce
        s  += __shfl_xor(s, o, 32);
        ss += __shfl_xor(ss, o, 32);
    }
    const int wave = threadIdx.x >> 5, lane = threadIdx.x & 31;
    if (lane == 0) { red[wave] = s; red[8 + wave] = ss; }
    __syncthreads();
    if (threadIdx.x == 0) {
        float ts = 0.f, tss = 0.f;
        #pragma unroll
        for (int i = 0; i < 8; ++i) { ts += red[i]; tss += red[8 + i]; }
        red[0] = ts; red[8] = tss;
    }
    __syncthreads();
    const float mu  = red[0] * (1.0f / Cc);
    const float var = red[8] * (1.0f / Cc) - mu * mu;
    const float inv = rsqrtf(var + 1e-5f);
    #pragma unroll
    for (int c = threadIdx.x; c < Cc; c += 256)
        y[row * Cc + c] = (bf16)((xr[c] - mu) * inv * w[c] + b[c]);
}

// ---------------------------------------------------------------------------
// Numerically-stable WKV scan. One thread per (b, c) channel; sequential in T.
// DIR=0 forward (writes 0.5*out), DIR=1 backward (accumulates 0.5*out).
// Loads/stores coalesce along C.
// ---------------------------------------------------------------------------
template <int DIR>
__global__ __launch_bounds__(256) void wkv_kernel(const float* __restrict__ kbuf,
                                                  const float* __restrict__ vbuf,
                                                  const float* __restrict__ decay,
                                                  const float* __restrict__ ubuf,
                                                  float* __restrict__ out) {
    const int g = blockIdx.x * 256 + threadIdx.x;  // 0 .. B*C-1
    const int c = g & (Cc - 1);
    const int b = g >> 10;                         // /Cc
    const float w  = -__expf(decay[c]);
    const float uu = ubuf[c];
    float a = 0.f, bb = 0.f, p = -1e38f;
    for (int t = 0; t < Tc; ++t) {
        const int tt = DIR ? (Tc - 1 - t) : t;
        const size_t idx = ((size_t)b * Tc + tt) * Cc + c;
        const float kk = kbuf[idx], vv = vbuf[idx];
        // emit output with bonus u on current token
        const float q  = fmaxf(p, uu + kk);
        const float e1 = __expf(p - q);
        const float e2 = __expf(uu + kk - q);
        const float o  = (e1 * a + e2 * vv) / (e1 * bb + e2);
        if (DIR) out[idx] += 0.5f * o;
        else     out[idx]  = 0.5f * o;
        // decay state and absorb current token
        const float q2 = fmaxf(p + w, kk);
        const float f1 = __expf(p + w - q2);
        const float f2 = __expf(kk - q2);
        a  = f1 * a + f2 * vv;
        bb = f1 * bb + f2;
        p  = q2;
    }
}

// ---------------------------------------------------------------------------
// rwkv = bf16( sigma_r * wkv )  elementwise
// ---------------------------------------------------------------------------
__global__ __launch_bounds__(256) void rmul_kernel(const float* __restrict__ r,
                                                   const float* __restrict__ wkv,
                                                   bf16* __restrict__ out, size_t n) {
    size_t i = (size_t)blockIdx.x * 256u + threadIdx.x;
    if (i < n) out[i] = (bf16)(r[i] * wkv[i]);
}

// ---------------------------------------------------------------------------
// Tiled WMMA GEMM:  Out[m,n] = epilogue( sum_k A[m,k] * W[n,k] )
//   A: [M,K] bf16 row-major (activations), W: [N,K] bf16 row-major (weight, used as W^T)
//   BM=BN=128, BK=32 (exactly one v_wmma_f32_16x16x32_bf16 K-step).
//   256 threads = 8 waves; wave (wm,wn) owns a 64x32 patch = 4x2 16x16 tiles.
// Fragment layouts per CDNA5 ISA 7.12.2:
//   A (16x32 bf16): lane<16 -> row=lane,   K in {half..half+7} U {16+half..16+half+7}, half = (lane>>4)*8
//   B (32x16 bf16): lane<16 -> col=lane&15, K in {kh..kh+7} U {16+kh..16+kh+7}, kh = (lane>>4)*8
//   C/D (16x16 f32): N = lane&15, M = vgpr + (lane>>4)*8
// ---------------------------------------------------------------------------
template <int EPI, typename OutT>
__global__ __launch_bounds__(256) void gemm_wmma_kernel(const bf16* __restrict__ A,
                                                        const bf16* __restrict__ W,
                                                        OutT* __restrict__ Out,
                                                        const float* __restrict__ res1,
                                                        const float* __restrict__ res2,
                                                        int M, int N, int K) {
    __shared__ __align__(16) bf16 As[128 * LDT];
    __shared__ __align__(16) bf16 Bs[128 * LDT];

    const int bm = blockIdx.y * 128;
    const int bn = blockIdx.x * 128;
    const int tid  = threadIdx.x;
    const int wave = tid >> 5;
    const int lane = tid & 31;
    const int wm = (wave & 1) * 64;   // wave's M offset within block tile
    const int wn = (wave >> 1) * 32;  // wave's N offset within block tile

    // tile-load mapping: thread t -> row t/2 (0..127), 16-elem half (t&1)
    const int lrow = tid >> 1;
    const int lcol = (tid & 1) * 16;

    v8f acc[4][2] = {};

    const int rl    = lane & 15;
    const int ahalf = (lane >> 4) * 8;    // A fragment K-half (interleaved 8s)
    const int bhalf = (lane >> 4) * 8;    // B fragment K-half (interleaved 8s)

    for (int k0 = 0; k0 < K; k0 += 32) {
        const bf16* ag = A + (size_t)(bm + lrow) * K + k0 + lcol;
        const bf16* bg = W + (size_t)(bn + lrow) * K + k0 + lcol;
        v8bf a0 = *(const v8bf*)(ag);
        v8bf a1 = *(const v8bf*)(ag + 8);
        v8bf b0 = *(const v8bf*)(bg);
        v8bf b1 = *(const v8bf*)(bg + 8);
        if (k0 + 32 < K) {                      // prefetch next k-tile (global_prefetch_b8)
            __builtin_prefetch(ag + 32, 0, 1);
            __builtin_prefetch(bg + 32, 0, 1);
        }
        __syncthreads();                        // previous iteration's reads done
        *(v8bf*)&As[lrow * LDT + lcol]     = a0;
        *(v8bf*)&As[lrow * LDT + lcol + 8] = a1;
        *(v8bf*)&Bs[lrow * LDT + lcol]     = b0;
        *(v8bf*)&Bs[lrow * LDT + lcol + 8] = b1;
        __syncthreads();

        v16bf af[4], bfr[2];
        #pragma unroll
        for (int mi = 0; mi < 4; ++mi) {
            const bf16* p = &As[(wm + mi * 16 + rl) * LDT + ahalf];
            v8bf lo = *(const v8bf*)(p);
            v8bf hi = *(const v8bf*)(p + 16);
            af[mi] = __builtin_shufflevector(lo, hi, 0, 1, 2, 3, 4, 5, 6, 7,
                                             8, 9, 10, 11, 12, 13, 14, 15);
        }
        #pragma unroll
        for (int ni = 0; ni < 2; ++ni) {
            const bf16* p = &Bs[(wn + ni * 16 + rl) * LDT + bhalf];
            v8bf lo = *(const v8bf*)(p);
            v8bf hi = *(const v8bf*)(p + 16);
            bfr[ni] = __builtin_shufflevector(lo, hi, 0, 1, 2, 3, 4, 5, 6, 7,
                                              8, 9, 10, 11, 12, 13, 14, 15);
        }
        #pragma unroll
        for (int mi = 0; mi < 4; ++mi)
            #pragma unroll
            for (int ni = 0; ni < 2; ++ni)
                acc[mi][ni] = __builtin_amdgcn_wmma_f32_16x16x32_bf16(
                    false, af[mi], false, bfr[ni], (short)0, acc[mi][ni], false, false);
    }

    // -------- fused epilogue + store (C/D layout: N=lane&15, M=vgpr+(lane>>4)*8)
    const int nlo = lane & 15;
    const int mhi = (lane >> 4) * 8;
    #pragma unroll
    for (int mi = 0; mi < 4; ++mi) {
        #pragma unroll
        for (int ni = 0; ni < 2; ++ni) {
            const int col   = bn + wn + ni * 16 + nlo;
            const int mbase = bm + wm + mi * 16 + mhi;
            #pragma unroll
            for (int r = 0; r < 8; ++r) {
                const size_t idx = (size_t)(mbase + r) * N + col;
                float v = acc[mi][ni][r];
                if constexpr (EPI == EPI_SIGMOID) {
                    ((float*)Out)[idx] = sigmoidf_fast(v);
                } else if constexpr (EPI == EPI_RESADD) {
                    ((float*)Out)[idx] = res1[idx] + v;
                } else if constexpr (EPI == EPI_RELUSQ) {
                    float t = fmaxf(v, 0.f);
                    ((bf16*)Out)[idx] = (bf16)(t * t);
                } else if constexpr (EPI == EPI_FINAL) {
                    ((float*)Out)[idx] = res1[idx] + res2[idx] * v;
                } else {
                    ((float*)Out)[idx] = v;
                }
            }
        }
    }
}

// ---------------------------------------------------------------------------
// host-side launcher
// ---------------------------------------------------------------------------
extern "C" void kernel_launch(void* const* d_in, const int* in_sizes, int n_in,
                              void* d_out, int out_size, void* d_ws, size_t ws_size,
                              hipStream_t stream) {
    (void)in_sizes; (void)n_in; (void)out_size; (void)ws_size;

    const float* x     = (const float*)d_in[0];
    const float* ln1_w = (const float*)d_in[1];
    const float* ln1_b = (const float*)d_in[2];
    const float* ln2_w = (const float*)d_in[3];
    const float* ln2_b = (const float*)d_in[4];
    const float* Wr    = (const float*)d_in[5];
    const float* Wk    = (const float*)d_in[6];
    const float* Wv    = (const float*)d_in[7];
    const float* Wo    = (const float*)d_in[8];
    const float* decay = (const float*)d_in[9];
    const float* u     = (const float*)d_in[10];
    const float* Wfk   = (const float*)d_in[11];
    const float* Wfv   = (const float*)d_in[12];
    const float* Wfr   = (const float*)d_in[13];
    float* out = (float*)d_out;

    const size_t CC  = (size_t)Cc * Cc;          // 1,048,576
    const size_t CC4 = 4 * CC;                   // 4,194,304
    const size_t ACT = (size_t)MT * Cc;          // 8,388,608

    // ---- workspace layout (~170 MB, buffers reused across disjoint lifetimes)
    char* p = (char*)d_ws;
    bf16* Wr_bf  = (bf16*)p;  p += CC  * sizeof(bf16);
    bf16* Wk_bf  = (bf16*)p;  p += CC  * sizeof(bf16);
    bf16* Wv_bf  = (bf16*)p;  p += CC  * sizeof(bf16);
    bf16* Wo_bf  = (bf16*)p;  p += CC  * sizeof(bf16);
    bf16* Wfr_bf = (bf16*)p;  p += CC  * sizeof(bf16);
    bf16* Wfk_bf = (bf16*)p;  p += CC4 * sizeof(bf16);
    bf16* Wfv_bf = (bf16*)p;  p += CC4 * sizeof(bf16);
    bf16* actA   = (bf16*)p;  p += ACT * sizeof(bf16);   // xn -> rwkv -> xn2
    float* bufR  = (float*)p; p += ACT * sizeof(float);  // sigma(r) -> sigma(fr)
    float* bufKV = (float*)p; p += 2 * ACT * sizeof(float); // k|v, later kk (bf16, 64MB)
    float* bufW  = (float*)p; p += ACT * sizeof(float);  // wkv -> x1
    float* kbuf  = bufKV;
    float* vbuf  = bufKV + ACT;
    bf16*  kkbuf = (bf16*)bufKV;

    dim3 blk(256);

    // 1) weights -> bf16
    f2bf_kernel<<<dim3((CC  + 255) / 256), blk, 0, stream>>>(Wr,  Wr_bf,  CC);
    f2bf_kernel<<<dim3((CC  + 255) / 256), blk, 0, stream>>>(Wk,  Wk_bf,  CC);
    f2bf_kernel<<<dim3((CC  + 255) / 256), blk, 0, stream>>>(Wv,  Wv_bf,  CC);
    f2bf_kernel<<<dim3((CC  + 255) / 256), blk, 0, stream>>>(Wo,  Wo_bf,  CC);
    f2bf_kernel<<<dim3((CC  + 255) / 256), blk, 0, stream>>>(Wfr, Wfr_bf, CC);
    f2bf_kernel<<<dim3((CC4 + 255) / 256), blk, 0, stream>>>(Wfk, Wfk_bf, CC4);
    f2bf_kernel<<<dim3((CC4 + 255) / 256), blk, 0, stream>>>(Wfv, Wfv_bf, CC4);

    // 2) LN1: x -> xn (bf16)
    ln_kernel<<<dim3(MT), blk, 0, stream>>>(x, ln1_w, ln1_b, actA);

    // 3) r = sigmoid(xn @ Wr^T); k = xn @ Wk^T; v = xn @ Wv^T
    dim3 gemmCC(Cc / 128, MT / 128);             // (8, 64)
    gemm_wmma_kernel<EPI_SIGMOID, float><<<gemmCC, blk, 0, stream>>>(
        actA, Wr_bf, bufR, nullptr, nullptr, MT, Cc, Cc);
    gemm_wmma_kernel<EPI_STORE, float><<<gemmCC, blk, 0, stream>>>(
        actA, Wk_bf, kbuf, nullptr, nullptr, MT, Cc, Cc);
    gemm_wmma_kernel<EPI_STORE, float><<<gemmCC, blk, 0, stream>>>(
        actA, Wv_bf, vbuf, nullptr, nullptr, MT, Cc, Cc);

    // 4) bidirectional WKV scan -> bufW = 0.5*(fwd + bwd)
    dim3 wkvGrid((Bc * Cc) / 256);               // 32 blocks, 1 thread per (b,c)
    wkv_kernel<0><<<wkvGrid, blk, 0, stream>>>(kbuf, vbuf, decay, u, bufW);
    wkv_kernel<1><<<wkvGrid, blk, 0, stream>>>(kbuf, vbuf, decay, u, bufW);

    // 5) rwkv = bf16(sigma_r * wkv)  (reuses actA)
    rmul_kernel<<<dim3(ACT / 256), blk, 0, stream>>>(bufR, bufW, actA, ACT);

    // 6) x1 = x + rwkv @ Wo^T  (reuses bufW)
    gemm_wmma_kernel<EPI_RESADD, float><<<gemmCC, blk, 0, stream>>>(
        actA, Wo_bf, bufW, x, nullptr, MT, Cc, Cc);

    // 7) LN2: x1 -> xn2 (bf16, reuses actA)
    ln_kernel<<<dim3(MT), blk, 0, stream>>>(bufW, ln2_w, ln2_b, actA);

    // 8) kk = bf16(relu(xn2 @ Wfk^T)^2)  [8192 x 4096], reuses bufKV
    dim3 gemmC4(4 * Cc / 128, MT / 128);         // (32, 64)
    gemm_wmma_kernel<EPI_RELUSQ, bf16><<<gemmC4, blk, 0, stream>>>(
        actA, Wfk_bf, kkbuf, nullptr, nullptr, MT, 4 * Cc, Cc);

    // 9) fr = sigmoid(xn2 @ Wfr^T)  (reuses bufR)
    gemm_wmma_kernel<EPI_SIGMOID, float><<<gemmCC, blk, 0, stream>>>(
        actA, Wfr_bf, bufR, nullptr, nullptr, MT, Cc, Cc);

    // 10) out = x1 + fr * (kk @ Wfv^T)   [K = 4096]
    gemm_wmma_kernel<EPI_FINAL, float><<<gemmCC, blk, 0, stream>>>(
        kkbuf, Wfv_bf, out, bufW, bufR, MT, Cc, 4 * Cc);
}
